// AtomicConvScore_47674136985686
// MI455X (gfx1250) — matrix operand order repaired and checked by reading, not compiled, not run on hardware
//
#include <hip/hip_runtime.h>
#include <hip/hip_bf16.h>

typedef __attribute__((ext_vector_type(16))) __bf16 v16bf;
typedef __attribute__((ext_vector_type(8)))  float  v8f;

#define T_TYPES 10
#define NTOT    2048     // atoms per batch across frag1(512)+frag2(512)+complex(1024)
#define NBATCH  32
#define CHUNKS  16       // tile-interleave factor for load balance

union FragU { uint4 u[2]; v16bf v; };

__device__ __forceinline__ unsigned short f2bf(float f) {
  unsigned u = __float_as_uint(f);
  unsigned r = ((u >> 16) & 1u) + 0x7FFFu;   // round-to-nearest-even
  return (unsigned short)((u + r) >> 16);
}
__device__ __forceinline__ float bf2f(unsigned short h) {
  return __uint_as_float(((unsigned)h) << 16);
}

// ---------------- init: zero counts + output ----------------
__global__ void init_k(int* counts, float* out) {
  int i = threadIdx.x;
  if (i < NBATCH * T_TYPES) counts[i] = 0;
  if (i < NBATCH) out[i] = 0.f;
}

// ---------------- bucket atoms by (batch, type) ----------------
__global__ void build_perm(const int* __restrict__ z1, const int* __restrict__ z2,
                           const int* __restrict__ zc, int* __restrict__ counts,
                           int* __restrict__ perm) {
  int idx = blockIdx.x * blockDim.x + threadIdx.x;   // 32*2048 threads
  int b = idx >> 11;
  int g = idx & (NTOT - 1);
  int z;
  if (g < 512)        z = z1[b * 512 + g];
  else if (g < 1024)  z = z2[b * 512 + (g - 512)];
  else                z = zc[b * 1024 + (g - 1024)];
  int pos = atomicAdd(&counts[b * T_TYPES + z], 1);
  perm[(b * T_TYPES + z) * NTOT + pos] = g;
}

// ---------------- repack f32 weights -> bf16 B-fragments ----------------
// Frag layout (per 32x16 K-tile x N-tile): lane L owns 8 dwords, dword j =
// pack(W[kb+2j][n], W[kb+2j+1][n]) with n = nt*16 + L%16, kb = kt*32 + 16*(L/16).
// Stored lane-contiguous: frag base + L*8 dwords  => 2x b128 per lane at load time.
__global__ void pack_weights(const float* __restrict__ W, unsigned int* __restrict__ dst,
                             int K, int H) {
  int idx = blockIdx.x * blockDim.x + threadIdx.x;
  int perType = (K * H) >> 1;
  if (idx >= T_TYPES * perType) return;
  int ty   = idx / perType;
  int rem  = idx - ty * perType;
  int frag = rem >> 8;          // 256 dwords per fragment
  int inF  = rem & 255;
  int lane = inF >> 3;
  int j    = inF & 7;
  int nt_count = H >> 4;
  int kt = frag / nt_count;
  int nt = frag - kt * nt_count;
  int n  = nt * 16 + (lane & 15);
  int kb = kt * 32 + ((lane >> 4) << 4) + 2 * j;
  const float* Wt = W + (size_t)ty * K * H;
  unsigned short lo = f2bf(Wt[(size_t)kb * H + n]);
  unsigned short hi = f2bf(Wt[(size_t)(kb + 1) * H + n]);
  dst[idx] = (unsigned)lo | ((unsigned)hi << 16);
}

// ---------------- one MLP layer on a 16-atom tile ----------------
// in : LDS [16][K] bf16 row-major.  A-frag: lane L -> m=L%16, s=L/16,
//      V0-7 = K in [kt*32+16s, +16) contiguous => two ds_load_b128.
// wf : frag-packed weights for this type/layer (global, L2-resident).
// oh : LDS [16][H] bf16 output (ReLU(acc + bias)).
template <int K, int H>
__device__ __forceinline__ void mlp_layer(const unsigned short* __restrict__ in,
                                          const unsigned int* __restrict__ wf,
                                          const float* __restrict__ bias,
                                          unsigned short* __restrict__ oh,
                                          int lane, int wv) {
  constexpr int NT = H / 16;
  constexpr int KT = K / 32;
  const int n = lane & 15;
  const int s = lane >> 4;
  const int m = n;
  for (int nt = wv; nt < NT; nt += 8) {
    v8f acc = {0.f, 0.f, 0.f, 0.f, 0.f, 0.f, 0.f, 0.f};
#pragma unroll
    for (int kt = 0; kt < KT; ++kt) {
      FragU A, Bf;
      A.u[0] = *reinterpret_cast<const uint4*>(in + m * K + kt * 32 + s * 16);
      A.u[1] = *reinterpret_cast<const uint4*>(in + m * K + kt * 32 + 8 + s * 16);
      const uint4* bp = reinterpret_cast<const uint4*>(wf) +
                        (size_t)(kt * NT + nt) * 64 + lane * 2;
      Bf.u[0] = bp[0];
      Bf.u[1] = bp[1];
      acc = __builtin_amdgcn_wmma_f32_16x16x32_bf16(
          false, A.v, false, Bf.v, (short)0, acc, false, false);
    }
    const int c  = nt * 16 + n;
    const float bv = bias[c];
#pragma unroll
    for (int r = 0; r < 8; ++r) {                  // D: VGPR r -> row r+8s, col c
      float v = acc[r] + bv;
      v = v > 0.f ? v : 0.f;
      oh[(r + 8 * s) * H + c] = f2bf(v);
    }
  }
}

// ---------------- main: per-(batch,type) compacted MLP ----------------
__global__ __launch_bounds__(256) void mlp_kernel(
    const float* __restrict__ x1, const float* __restrict__ x2,
    const float* __restrict__ xc,
    const int* __restrict__ counts, const int* __restrict__ perm,
    const unsigned int* __restrict__ w0f, const unsigned int* __restrict__ w1f,
    const unsigned int* __restrict__ w2f,
    const float* __restrict__ b0, const float* __restrict__ b1,
    const float* __restrict__ b2,
    const float* __restrict__ wout, const float* __restrict__ bout,
    float* __restrict__ out) {
  const int blk = blockIdx.x;
  const int b   = blk / (T_TYPES * CHUNKS);
  const int rem = blk % (T_TYPES * CHUNKS);
  const int t   = rem / CHUNKS;
  const int ch  = rem % CHUNKS;
  const int cnt = counts[b * T_TYPES + t];
  if (ch * 16 >= cnt) return;                       // uniform exit per block

  __shared__ __align__(16) unsigned short xs[16][256];
  __shared__ __align__(16) unsigned short h0[16][256];
  __shared__ __align__(16) unsigned short h1[16][256];
  __shared__ __align__(16) unsigned short h2[16][128];
  __shared__ const float* rowp[16];
  __shared__ float sgn[16];
  __shared__ float part[16][8];
  __shared__ float redn[16];
  __shared__ float blockAcc;

  const int tid  = threadIdx.x;
  const int lane = tid & 31;
  const int wv   = tid >> 5;
  if (tid == 0) blockAcc = 0.f;

  const unsigned int* w0t = w0f + (size_t)t * 32768;   // 256*256/2 dwords
  const unsigned int* w1t = w1f + (size_t)t * 32768;
  const unsigned int* w2t = w2f + (size_t)t * 16384;   // 256*128/2 dwords
  const float* b0t = b0 + t * 256;
  const float* b1t = b1 + t * 256;
  const float* b2t = b2 + t * 128;
  const float* wot = wout + t * 128;
  const float  bot = bout[t];

  for (int base = ch * 16; base < cnt; base += 16 * CHUNKS) {
    // ---- gather row pointers + signs (pad rows -> sign 0) ----
    if (tid < 16) {
      int i = base + tid;
      if (i < cnt) {
        int g = perm[(b * T_TYPES + t) * NTOT + i];
        if (g < 512)        { rowp[tid] = x1 + ((size_t)b * 512 + g) * 256;           sgn[tid] = -1.f; }
        else if (g < 1024)  { rowp[tid] = x2 + ((size_t)b * 512 + (g - 512)) * 256;   sgn[tid] = -1.f; }
        else                { rowp[tid] = xc + ((size_t)b * 1024 + (g - 1024)) * 256; sgn[tid] =  1.f; }
      } else { rowp[tid] = x1 + (size_t)b * 512 * 256; sgn[tid] = 0.f; }
    }
    __syncthreads();

    // ---- stage activations f32 -> bf16 in LDS ----
    {
      int r  = tid >> 4;
      int c0 = tid & 15;
      const float* p = rowp[r];
#pragma unroll
      for (int i = 0; i < 16; ++i) {
        int c = c0 + i * 16;
        xs[r][c] = f2bf(p[c]);
      }
    }
    __syncthreads();

    mlp_layer<256, 256>(&xs[0][0], w0t, b0t, &h0[0][0], lane, wv);
    __syncthreads();
    mlp_layer<256, 256>(&h0[0][0], w1t, b1t, &h1[0][0], lane, wv);
    __syncthreads();
    mlp_layer<256, 128>(&h1[0][0], w2t, b2t, &h2[0][0], lane, wv);
    __syncthreads();

    // ---- final dot: out[m] = sum_c h2[m][c] * wout[c] + bout ----
    if (tid < 128) {
      int m = tid >> 3, q = tid & 7;
      float s = 0.f;
#pragma unroll
      for (int c = q * 16; c < q * 16 + 16; ++c) s += bf2f(h2[m][c]) * wot[c];
      part[m][q] = s;
    }
    __syncthreads();
    if (tid < 16) {
      float s = 0.f;
#pragma unroll
      for (int q = 0; q < 8; ++q) s += part[tid][q];
      redn[tid] = sgn[tid] * (s + bot);
    }
    __syncthreads();
    if (tid == 0) {
      float s = 0.f;
#pragma unroll
      for (int i = 0; i < 16; ++i) s += redn[i];
      blockAcc += s;
    }
    __syncthreads();
  }
  if (tid == 0) atomicAdd(&out[b], blockAcc);
}

// ---------------- host-side launch ----------------
extern "C" void kernel_launch(void* const* d_in, const int* in_sizes, int n_in,
                              void* d_out, int out_size, void* d_ws, size_t ws_size,
                              hipStream_t stream) {
  const float* x1 = (const float*)d_in[0];
  const float* x2 = (const float*)d_in[1];
  const float* xc = (const float*)d_in[2];
  const int*   z1 = (const int*)d_in[3];
  const int*   z2 = (const int*)d_in[4];
  const int*   zc = (const int*)d_in[5];
  // d_in[6] = atom_types == arange(10): identity, unused
  const float* W0 = (const float*)d_in[7];
  const float* b0 = (const float*)d_in[8];
  const float* W1 = (const float*)d_in[9];
  const float* b1 = (const float*)d_in[10];
  const float* W2 = (const float*)d_in[11];
  const float* b2 = (const float*)d_in[12];
  const float* Wo = (const float*)d_in[13];
  const float* bo = (const float*)d_in[14];
  float* out = (float*)d_out;

  // workspace layout (bytes)
  char* ws = (char*)d_ws;
  int*  counts = (int*)(ws);                                   // 1280 B (pad 4K)
  int*  perm   = (int*)(ws + 4096);                            // 32*10*2048*4 = 2,621,440
  unsigned int* w0f = (unsigned int*)(ws + 4096 + 2621440);    // 1,310,720
  unsigned int* w1f = (unsigned int*)(ws + 4096 + 2621440 + 1310720);
  unsigned int* w2f = (unsigned int*)(ws + 4096 + 2621440 + 2 * 1310720);

  init_k<<<1, 384, 0, stream>>>(counts, out);
  build_perm<<<(NBATCH * NTOT) / 256, 256, 0, stream>>>(z1, z2, zc, counts, perm);
  pack_weights<<<1280, 256, 0, stream>>>(W0, w0f, 256, 256);
  pack_weights<<<1280, 256, 0, stream>>>(W1, w1f, 256, 256);
  pack_weights<<<640, 256, 0, stream>>>(W2, w2f, 256, 128);
  mlp_kernel<<<NBATCH * T_TYPES * CHUNKS, 256, 0, stream>>>(
      x1, x2, xc, counts, perm, w0f, w1f, w2f, b0, b1, b2, Wo, bo, out);
}
